// GCRec_61795989454855
// MI455X (gfx1250) — compile-verified, matrix operand order checked by reading, not run on hardware
//
#include <hip/hip_runtime.h>
#include <hip/hip_bf16.h>

typedef __attribute__((ext_vector_type(16))) _Float16 v16h;
typedef __attribute__((ext_vector_type(8)))  _Float16 v8h;
typedef __attribute__((ext_vector_type(8)))  float    v8f;

constexpr int NUSERS = 60000;
constexpr int NITEMS = 40000;
constexpr int NNODES = 100000;
constexpr int NNZ    = 2000000;
constexpr int DIM    = 64;
constexpr int BATCH  = 4096;
constexpr float INV_T = 5.0f;     // 1 / 0.2; also the fixed logsumexp shift
constexpr float REG_L = 1e-4f;
constexpr float SSL_L = 0.1f;

// Guaranteed hardware global_atomic_add_f32 (no CAS loop): relaxed, agent scope.
__device__ __forceinline__ void atomicAddF32(float* p, float v) {
    __hip_atomic_fetch_add(p, v, __ATOMIC_RELAXED, __HIP_MEMORY_SCOPE_AGENT);
}

// ---------------------------------------------------------------------------
// init: x = concat(user_table, item_table); acc = 0; loss accumulators = 0
// ---------------------------------------------------------------------------
__global__ void k_init(const float* __restrict__ ut, const float* __restrict__ it,
                       float* __restrict__ x, float* __restrict__ acc,
                       float* __restrict__ accs) {
    int i = blockIdx.x * blockDim.x + threadIdx.x;
    if (i < NNODES * DIM) {
        float v = (i < NUSERS * DIM) ? ut[i] : it[i - NUSERS * DIM];
        x[i]   = v;
        acc[i] = 0.0f;
    }
    if (blockIdx.x == 0 && threadIdx.x < 8) accs[threadIdx.x] = 0.0f;
}

__global__ void k_zero(float* __restrict__ y) {
    int i = blockIdx.x * blockDim.x + threadIdx.x;
    if (i < NNODES * DIM) y[i] = 0.0f;
}

// ---------------------------------------------------------------------------
// SpMM: one wave32 per edge; lane moves float2 -> full 256B coalesced row.
// y[row] += val * x[col]   via hardware global_atomic_add_f32
// ---------------------------------------------------------------------------
__global__ void k_spmm(const float* __restrict__ x, float* __restrict__ y,
                       const int* __restrict__ er, const int* __restrict__ ec,
                       const float* __restrict__ ev) {
    int gid  = blockIdx.x * blockDim.x + threadIdx.x;
    int e    = gid >> 5;
    int lane = gid & 31;
    if (e >= NNZ) return;
    int c   = ec[e];
    int r   = er[e];
    float v = ev[e];
    float2 d = ((const float2*)(x + (size_t)c * DIM))[lane];
    float* yp = y + (size_t)r * DIM + 2 * lane;
    atomicAddF32(yp,     v * d.x);
    atomicAddF32(yp + 1, v * d.y);
}

__global__ void k_accum(float* __restrict__ acc, const float* __restrict__ y) {
    int i = blockIdx.x * blockDim.x + threadIdx.x;
    if (i < NNODES * DIM) acc[i] += y[i];
}

// ---------------------------------------------------------------------------
// Gather batch rows of acc, L2-normalize (the /3 layer-mean cancels:
// (a/3)/(||a||/3 + 1e-8) == a/(||a|| + 3e-8)), write f16 operand matrix,
// and accumulate -pos = -(v.v)/T into the SSL accumulator.
// One wave per batch row.
// ---------------------------------------------------------------------------
__global__ void k_gather_norm(const float* __restrict__ acc, const int* __restrict__ idx,
                              int offset, _Float16* __restrict__ outh,
                              float* __restrict__ ssl_acc) {
    int gid  = blockIdx.x * blockDim.x + threadIdx.x;
    int b    = gid >> 5;
    int lane = gid & 31;
    if (b >= BATCH) return;
    const float* row = acc + (size_t)(idx[b] + offset) * DIM;
    float2 d = ((const float2*)row)[lane];
    float ss = d.x * d.x + d.y * d.y;
#pragma unroll
    for (int m = 16; m > 0; m >>= 1) ss += __shfl_xor(ss, m, 32);
    float sc = 1.0f / (sqrtf(ss) + 3e-8f);
    outh[(size_t)b * DIM + 2 * lane]     = (_Float16)(d.x * sc);
    outh[(size_t)b * DIM + 2 * lane + 1] = (_Float16)(d.y * sc);
    if (lane == 0) atomicAddF32(ssl_acc, -(ss * sc * sc) * INV_T);
}

// ---------------------------------------------------------------------------
// InfoNCE logits S = Vh * Vh^T / T with fixed-shift streaming logsumexp.
// Inputs are L2-normalized, so every logit is in [-1/T, +1/T] = [-5, 5].
// Hence   logsumexp_r = INV_T + log( sum_j exp(l_rj - INV_T) )
// is exact, overflow-free (exp(.) <= 1), and needs NO running max:
// the WMMA tail is one v_exp + one add per element.
//
// Each wave owns one 16-row tile; loops over 256 column tiles; K=64 done as
// two chained v_wmma_f32_16x16x32_f16. The 64MB logits matrix never exists.
// A-operand layout (16-bit A 16x32): lanes 0-15 = M, K {0-7,16-23};
// lanes 16-31 = same M, K {8-15,24-31}. B uses the symmetric layout, so for
// V*V^T both operands load row tiles with the identical per-lane pattern.
// C layout: lane L, vgpr v -> (M = v + 8*(L>>4), N = L&15).
// ---------------------------------------------------------------------------
__global__ void __launch_bounds__(256) k_infonce(const _Float16* __restrict__ Vh,
                                                 float* __restrict__ ssl_acc) {
    int wave = (blockIdx.x * blockDim.x + threadIdx.x) >> 5;   // row-tile id 0..255
    int lane = threadIdx.x & 31;
    int mrow = wave * 16 + (lane & 15);
    int khi  = (lane >> 4) * 8;

    const _Float16* ar = Vh + (size_t)mrow * DIM;
    v8h a0lo = *(const v8h*)(ar +  0 + khi);
    v8h a0hi = *(const v8h*)(ar + 16 + khi);
    v8h a1lo = *(const v8h*)(ar + 32 + khi);
    v8h a1hi = *(const v8h*)(ar + 48 + khi);
    v16h a0 = __builtin_shufflevector(a0lo, a0hi, 0,1,2,3,4,5,6,7,8,9,10,11,12,13,14,15);
    v16h a1 = __builtin_shufflevector(a1lo, a1hi, 0,1,2,3,4,5,6,7,8,9,10,11,12,13,14,15);

    float s[8];
#pragma unroll
    for (int v = 0; v < 8; ++v) s[v] = 0.0f;

    for (int j = 0; j < BATCH / 16; ++j) {
        const _Float16* br = Vh + (size_t)(j * 16 + (lane & 15)) * DIM;
        v8h b0lo = *(const v8h*)(br +  0 + khi);
        v8h b0hi = *(const v8h*)(br + 16 + khi);
        v8h b1lo = *(const v8h*)(br + 32 + khi);
        v8h b1hi = *(const v8h*)(br + 48 + khi);
        v16h b0 = __builtin_shufflevector(b0lo, b0hi, 0,1,2,3,4,5,6,7,8,9,10,11,12,13,14,15);
        v16h b1 = __builtin_shufflevector(b1lo, b1hi, 0,1,2,3,4,5,6,7,8,9,10,11,12,13,14,15);

        v8f c = {};
        c = __builtin_amdgcn_wmma_f32_16x16x32_f16(false, a0, false, b0, (short)0, c, false, false);
        c = __builtin_amdgcn_wmma_f32_16x16x32_f16(false, a1, false, b1, (short)0, c, false, false);

#pragma unroll
        for (int v = 0; v < 8; ++v) {
            // l - shift = c*INV_T - INV_T = (c - 1) * INV_T   (always <= 0)
            s[v] += __expf((c[v] - 1.0f) * INV_T);
        }
    }
    // sum across the 16 lanes that share each row (lane halves hold disjoint rows)
#pragma unroll
    for (int mask = 1; mask < 16; mask <<= 1) {
#pragma unroll
        for (int v = 0; v < 8; ++v) s[v] += __shfl_xor(s[v], mask, 32);
    }
    if ((lane & 15) == 0) {
        float sum = 0.0f;
#pragma unroll
        for (int v = 0; v < 8; ++v) sum += INV_T + __logf(s[v]);
        atomicAddF32(ssl_acc, sum);   // sum of logsumexp for 8 rows
    }
}

// ---------------------------------------------------------------------------
// BPR: softplus(neg - pos) over final embeddings; final = acc/3 folds into 1/9
// on the dot products. One wave per batch element.
// ---------------------------------------------------------------------------
__global__ void k_bpr(const float* __restrict__ acc, const int* __restrict__ u,
                      const int* __restrict__ p, const int* __restrict__ n,
                      float* __restrict__ bpr_acc) {
    int gid  = blockIdx.x * blockDim.x + threadIdx.x;
    int b    = gid >> 5;
    int lane = gid & 31;
    if (b >= BATCH) return;
    float2 ue = ((const float2*)(acc + (size_t)u[b] * DIM))[lane];
    float2 pe = ((const float2*)(acc + (size_t)(NUSERS + p[b]) * DIM))[lane];
    float2 ne = ((const float2*)(acc + (size_t)(NUSERS + n[b]) * DIM))[lane];
    float dd = (ue.x * ne.x + ue.y * ne.y) - (ue.x * pe.x + ue.y * pe.y);
#pragma unroll
    for (int m = 16; m > 0; m >>= 1) dd += __shfl_xor(dd, m, 32);
    if (lane == 0) {
        float x  = dd * (1.0f / 9.0f);
        float sp = fmaxf(x, 0.0f) + log1pf(__expf(-fabsf(x)));
        atomicAddF32(bpr_acc, sp);
    }
}

// ---------------------------------------------------------------------------
// Reg: 0.5 * sum of squares of ego embeddings / B (lambda applied at the end).
// ---------------------------------------------------------------------------
__global__ void k_reg(const float* __restrict__ ut, const float* __restrict__ it,
                      const int* __restrict__ u, const int* __restrict__ p,
                      const int* __restrict__ n, float* __restrict__ reg_acc) {
    int gid  = blockIdx.x * blockDim.x + threadIdx.x;
    int b    = gid >> 5;
    int lane = gid & 31;
    if (b >= BATCH) return;
    float2 a = ((const float2*)(ut + (size_t)u[b] * DIM))[lane];
    float2 c = ((const float2*)(it + (size_t)p[b] * DIM))[lane];
    float2 d = ((const float2*)(it + (size_t)n[b] * DIM))[lane];
    float ss = a.x*a.x + a.y*a.y + c.x*c.x + c.y*c.y + d.x*d.x + d.y*d.y;
#pragma unroll
    for (int m = 16; m > 0; m >>= 1) ss += __shfl_xor(ss, m, 32);
    if (lane == 0) atomicAddF32(reg_acc, ss);
}

__global__ void k_final(const float* __restrict__ accs, float* __restrict__ out) {
    if (blockIdx.x == 0 && threadIdx.x == 0) {
        out[0] = accs[0] / (float)BATCH;                       // bpr
        out[1] = REG_L * 0.5f * accs[1] / (float)BATCH;        // reg
        out[2] = SSL_L * (accs[2] + accs[3]) / (float)BATCH;   // ssl (lse - pos sums)
    }
}

// ---------------------------------------------------------------------------
extern "C" void kernel_launch(void* const* d_in, const int* in_sizes, int n_in,
                              void* d_out, int out_size, void* d_ws, size_t ws_size,
                              hipStream_t stream) {
    const float* user_table = (const float*)d_in[0];
    const float* item_table = (const float*)d_in[1];
    const float* edge_val   = (const float*)d_in[2];
    const int*   user       = (const int*)d_in[3];
    const int*   positive   = (const int*)d_in[4];
    const int*   negative   = (const int*)d_in[5];
    const int*   edge_row   = (const int*)d_in[6];
    const int*   edge_col   = (const int*)d_in[7];
    float* out = (float*)d_out;

    char*  ws  = (char*)d_ws;
    size_t off = 0;
    auto carve = [&](size_t bytes) -> char* {
        char* p = ws + off;
        off += (bytes + 255) & ~(size_t)255;
        return p;
    };
    float*    bufA = (float*)   carve(sizeof(float)    * NNODES * DIM);
    float*    bufB = (float*)   carve(sizeof(float)    * NNODES * DIM);
    float*    acc  = (float*)   carve(sizeof(float)    * NNODES * DIM);
    _Float16* uh   = (_Float16*)carve(sizeof(_Float16) * BATCH  * DIM);
    _Float16* ih   = (_Float16*)carve(sizeof(_Float16) * BATCH  * DIM);
    float*    accs = (float*)   carve(256);   // [0]=bpr [1]=reg [2]=ssl_u [3]=ssl_i

    const int TB    = 256;
    const int gridN = (NNODES * DIM + TB - 1) / TB;
    const int gridE = (NNZ * 32 + TB - 1) / TB;   // one wave per edge
    const int gridB = (BATCH * 32) / TB;          // one wave per batch row

    k_init<<<gridN, TB, 0, stream>>>(user_table, item_table, bufA, acc, accs);

    // layer 0: bufA -> bufB
    k_zero <<<gridN, TB, 0, stream>>>(bufB);
    k_spmm <<<gridE, TB, 0, stream>>>(bufA, bufB, edge_row, edge_col, edge_val);
    k_accum<<<gridN, TB, 0, stream>>>(acc, bufB);
    // layer 1: bufB -> bufA
    k_zero <<<gridN, TB, 0, stream>>>(bufA);
    k_spmm <<<gridE, TB, 0, stream>>>(bufB, bufA, edge_row, edge_col, edge_val);
    k_accum<<<gridN, TB, 0, stream>>>(acc, bufA);
    // layer 2: bufA -> bufB
    k_zero <<<gridN, TB, 0, stream>>>(bufB);
    k_spmm <<<gridE, TB, 0, stream>>>(bufA, bufB, edge_row, edge_col, edge_val);
    k_accum<<<gridN, TB, 0, stream>>>(acc, bufB);

    k_gather_norm<<<gridB, TB, 0, stream>>>(acc, user,     0,      uh, accs + 2);
    k_gather_norm<<<gridB, TB, 0, stream>>>(acc, positive, NUSERS, ih, accs + 3);
    k_infonce<<<(BATCH / 16) / 8, TB, 0, stream>>>(uh, accs + 2);
    k_infonce<<<(BATCH / 16) / 8, TB, 0, stream>>>(ih, accs + 3);

    k_bpr<<<gridB, TB, 0, stream>>>(acc, user, positive, negative, accs + 0);
    k_reg<<<gridB, TB, 0, stream>>>(user_table, item_table, user, positive, negative, accs + 1);

    k_final<<<1, 32, 0, stream>>>(accs, out);
}
